// MultiheadCorrelation_64957085385312
// MI455X (gfx1250) — matrix operand order repaired and checked by reading, not compiled, not run on hardware
//
#include <hip/hip_runtime.h>
#include <hip/hip_bf16.h>
#include <stdint.h>

// ---------------------------------------------------------------------------
// MI455X (gfx1250) multihead-correlation attention, bf16 WMMA pipeline.
//   T=2048, B=2, D=1024, H=16, hd=64, S=T=2048, rows = T*B = 4096
// Stage 1: QKV GEMM (bf16 WMMA, fp32 acc) + bias + per-head row centering,
//          writes qc,kc [BH,S,64] bf16 and v transposed vT [BH,64,S] bf16.
// Stage 1.5: per row, d = qc.kc; scale both rows by 1/sqrt(d) IN PLACE
//          (NaN when d<0 — reproduces reference sqrt(diag(q k^T)) + NaN->0).
// Stage 2: fused attention: S = qn @ kn^T via WMMA, clip to [-1,1], NaN->0,
//          exp (corr bounded, no running max needed), running row sums,
//          P@V via WMMA, normalize, store fp32 [B,H,S,hd].
// Stage-2 staging: DOUBLE-BUFFERED async global->LDS DMA (ASYNCcnt path) —
// tile i+1 streams into the other LDS buffer while tile i's WMMAs run.
// ---------------------------------------------------------------------------

#define NHEAD   16
#define S_LEN   2048
#define BATCH   2
#define DMODEL  1024
#define HD      64
#define BH      (BATCH * NHEAD)   // 32
#define ROWS    (S_LEN * BATCH)   // 4096 tokens
#define NKT     (S_LEN / 64)      // 32 k-tiles

#if __has_builtin(__builtin_amdgcn_global_load_async_to_lds_b128) && \
    __has_builtin(__builtin_amdgcn_s_wait_asynccnt)
#define USE_ASYNC_LDS 1
#else
#define USE_ASYNC_LDS 0
#endif

typedef __attribute__((ext_vector_type(16))) __bf16 v16bf;
typedef __attribute__((ext_vector_type(8)))  float  v8f;
typedef int v4i __attribute__((__vector_size__(4 * sizeof(int))));
typedef __attribute__((address_space(1))) v4i* gv4i_ptr;   // global (AS1)
typedef __attribute__((address_space(3))) v4i* lv4i_ptr;   // LDS (AS3)

union Frag {
    v16bf          v;
    uint4          q[2];
    unsigned int   u[8];
    unsigned short s[16];
};

// Branchless conversions: let clang emit the native cvt (RNE, NaN-safe).
__device__ __forceinline__ unsigned short f2bf(float f) {
    union { __bf16 h; unsigned short s; } x;
    x.h = (__bf16)f;
    return x.s;
}
__device__ __forceinline__ float bf2f(unsigned short h) {
    union { unsigned short s; __bf16 h; } x;
    x.s = h;
    return (float)x.h;
}
__device__ __forceinline__ unsigned pk2(float a, float b) {
    return (unsigned)f2bf(a) | ((unsigned)f2bf(b) << 16);
}
__device__ __forceinline__ unsigned scale_pk(unsigned pk, float s) {
    float lo = bf2f((unsigned short)(pk & 0xffffu)) * s;
    float hi = bf2f((unsigned short)(pk >> 16)) * s;
    return pk2(lo, hi);
}

// ---------------------------------------------------------------------------
// Stage 1: QKV projection GEMM. grid = (ROWS/64, 3*DMODEL/64), block = 128.
// Each block: 64x64 tile; N-tile is exactly one head slice of q, k or v.
// ---------------------------------------------------------------------------
__global__ __launch_bounds__(128)
void qkv_gemm_kernel(const float* __restrict__ X,      // [4096,1024]
                     const float* __restrict__ W,      // [3072,1024]
                     const float* __restrict__ bias,   // [3072]
                     unsigned short* __restrict__ qc,  // [BH,S,64] bf16 centered
                     unsigned short* __restrict__ kc,  // [BH,S,64] bf16 centered
                     unsigned short* __restrict__ vT)  // [BH,64,S] bf16
{
    __shared__ unsigned short xT[64 * 64];
    __shared__ unsigned short wT[64 * 64];

    const int tid  = threadIdx.x;
    const int wave = tid >> 5;
    const int lane = tid & 31;
    const int ln   = lane & 15;
    const int hf   = lane >> 4;
    const int m0   = blockIdx.x * 64;
    const int n0   = blockIdx.y * 64;

    const v8f zero = {0.f, 0.f, 0.f, 0.f, 0.f, 0.f, 0.f, 0.f};
    v8f acc[4] = {zero, zero, zero, zero};

    for (int k0 = 0; k0 < DMODEL; k0 += 64) {
        // cooperative fp32 -> bf16 staging (packed dword stores)
        for (int j = 0; j < 8; ++j) {
            int e4  = tid + j * 128;          // float4 id, 1024 total
            int row = e4 >> 4;
            int col = (e4 & 15) << 2;
            float4 xv = *(const float4*)(X + (size_t)(m0 + row) * DMODEL + k0 + col);
            float4 wv = *(const float4*)(W + (size_t)(n0 + row) * DMODEL + k0 + col);
            uint2 xp, wp;
            xp.x = pk2(xv.x, xv.y); xp.y = pk2(xv.z, xv.w);
            wp.x = pk2(wv.x, wv.y); wp.y = pk2(wv.z, wv.w);
            *(uint2*)&xT[row * 64 + col] = xp;
            *(uint2*)&wT[row * 64 + col] = wp;
        }
        __syncthreads();

        // A fragments: rows wave*16 + (lane&15); K = e + 8*hf + 8*(e>>3)
        Frag a[2];
        const int arow = wave * 16 + ln;
        for (int ks = 0; ks < 2; ++ks) {
            a[ks].q[0] = *(const uint4*)&xT[arow * 64 + ks * 32 + 8 * hf];
            a[ks].q[1] = *(const uint4*)&xT[arow * 64 + ks * 32 + 8 * hf + 16];
        }
        for (int ns = 0; ns < 4; ++ns) {
            for (int ks = 0; ks < 2; ++ks) {
                Frag b;  // B fragment: row n' = ns*16 + ln; K = e + 8*hf + 8*(e>>3)
                b.q[0] = *(const uint4*)&wT[(ns * 16 + ln) * 64 + ks * 32 + 8 * hf];
                b.q[1] = *(const uint4*)&wT[(ns * 16 + ln) * 64 + ks * 32 + 8 * hf + 16];
                acc[ns] = __builtin_amdgcn_wmma_f32_16x16x32_bf16(
                    false, a[ks].v, false, b.v, (short)0, acc[ns], false, false);
            }
        }
        __syncthreads();
    }

    // bias
    const int region = n0 >> 10;             // 0:q 1:k 2:v
    const int h      = (n0 & 1023) >> 6;     // head index
    float bj[4];
    for (int ns = 0; ns < 4; ++ns) bj[ns] = bias[n0 + ns * 16 + ln];
    for (int ns = 0; ns < 4; ++ns)
        for (int r = 0; r < 8; ++r) acc[ns][r] += bj[ns];

    unsigned short* dstc = (region == 0) ? qc : kc;
    for (int r = 0; r < 8; ++r) {
        int i  = m0 + wave * 16 + r + 8 * hf;    // token row = t*B + b
        int bb = i & (BATCH - 1);
        int t  = i >> 1;
        int bh = bb * NHEAD + h;
        if (region < 2) {
            // row mean over the 64 head columns (4 nsub x 16 lanes)
            float s = acc[0][r] + acc[1][r] + acc[2][r] + acc[3][r];
            s += __shfl_xor(s, 1, 32);
            s += __shfl_xor(s, 2, 32);
            s += __shfl_xor(s, 4, 32);
            s += __shfl_xor(s, 8, 32);
            float mean = s * (1.0f / 64.0f);
            unsigned short* base = dstc + ((size_t)(bh * S_LEN + t)) * HD;
            for (int ns = 0; ns < 4; ++ns)
                base[ns * 16 + ln] = f2bf(acc[ns][r] - mean);
        } else {
            for (int ns = 0; ns < 4; ++ns)
                vT[((size_t)(bh * HD + ns * 16 + ln)) * S_LEN + t] = f2bf(acc[ns][r]);
        }
    }
}

// ---------------------------------------------------------------------------
// Stage 1.5: d = qc.kc per row; scale both rows in place by 1/sqrt(d).
// One thread per (b,h,s) row.
// ---------------------------------------------------------------------------
__global__ __launch_bounds__(256)
void normalize_kernel(unsigned short* __restrict__ qc,
                      unsigned short* __restrict__ kc)
{
    int row = blockIdx.x * blockDim.x + threadIdx.x;   // 0 .. BH*S-1
    uint4* qp = (uint4*)(qc + (size_t)row * HD);
    uint4* kp = (uint4*)(kc + (size_t)row * HD);
    uint4 qv[8], kv[8];
    float d = 0.f;
    for (int j = 0; j < 8; ++j) {
        qv[j] = qp[j];
        kv[j] = kp[j];
        const unsigned* qa = (const unsigned*)&qv[j];
        const unsigned* ka = (const unsigned*)&kv[j];
        for (int c = 0; c < 4; ++c) {
            d += bf2f((unsigned short)(qa[c] & 0xffffu)) * bf2f((unsigned short)(ka[c] & 0xffffu));
            d += bf2f((unsigned short)(qa[c] >> 16))     * bf2f((unsigned short)(ka[c] >> 16));
        }
    }
    float inv = 1.0f / sqrtf(d);   // NaN when d<0 -> matches reference NaN->0 path
    for (int j = 0; j < 8; ++j) {
        unsigned* qa = (unsigned*)&qv[j];
        unsigned* ka = (unsigned*)&kv[j];
        for (int c = 0; c < 4; ++c) {
            qa[c] = scale_pk(qa[c], inv);
            ka[c] = scale_pk(ka[c], inv);
        }
        qp[j] = qv[j];
        kp[j] = kv[j];
    }
}

// ---------------------------------------------------------------------------
// Stage 2: fused correlation-attention. grid = BH*(S/64), block = 128.
// Each wave owns 16 q rows; corr in [-1,1] so exp needs no running max.
// Double-buffered async global->LDS staging overlaps DMA with WMMA.
// ---------------------------------------------------------------------------
__device__ __forceinline__
void stage_tile(const unsigned short* __restrict__ kn,
                const unsigned short* __restrict__ vT,
                unsigned short* kbuf, unsigned short* vbuf,
                int bh, int kt, int tid)
{
    for (int j = 0; j < 4; ++j) {
        int e4  = tid + j * 128;            // 16-byte unit id, 512 total
        int row = e4 >> 3;
        int col = (e4 & 7) << 3;
#if USE_ASYNC_LDS
        __builtin_amdgcn_global_load_async_to_lds_b128(
            (gv4i_ptr)(kn + ((size_t)(bh * S_LEN + kt + row)) * HD + col),
            (lv4i_ptr)&kbuf[row * 64 + col], 0, 0);
        __builtin_amdgcn_global_load_async_to_lds_b128(
            (gv4i_ptr)(vT + ((size_t)(bh * HD + row)) * S_LEN + kt + col),
            (lv4i_ptr)&vbuf[row * 64 + col], 0, 0);
#else
        *(uint4*)&kbuf[row * 64 + col] =
            *(const uint4*)(kn + ((size_t)(bh * S_LEN + kt + row)) * HD + col);
        *(uint4*)&vbuf[row * 64 + col] =
            *(const uint4*)(vT + ((size_t)(bh * HD + row)) * S_LEN + kt + col);
#endif
    }
}

__global__ __launch_bounds__(128)
void attn_kernel(const unsigned short* __restrict__ qn,  // [BH,S,64] normalized
                 const unsigned short* __restrict__ kn,  // [BH,S,64] normalized
                 const unsigned short* __restrict__ vT,  // [BH,64,S]
                 float* __restrict__ out)                // [B,H,S,hd]
{
    __shared__ unsigned short kT[2][64 * 64];    // kn tiles, row-major [s][hd]
    __shared__ unsigned short vTile[2][64 * 64]; // v^T tiles, [hd][s]
    __shared__ unsigned short pT[4][16 * 64];    // per-wave P scratch

    const int tid  = threadIdx.x;
    const int wave = tid >> 5;
    const int lane = tid & 31;
    const int ln   = lane & 15;
    const int hf   = lane >> 4;
    const int bh   = blockIdx.x >> 5;            // 32 q-blocks per (b,h)
    const int q0   = (blockIdx.x & 31) * 64;

    // q A-fragments, loaded once (already normalized)
    const int qrow = q0 + wave * 16 + ln;
    Frag aq[2];
    {
        const unsigned short* qbase = qn + ((size_t)(bh * S_LEN + qrow)) * HD;
        for (int ks = 0; ks < 2; ++ks) {
            aq[ks].q[0] = *(const uint4*)(qbase + ks * 32 + 8 * hf);
            aq[ks].q[1] = *(const uint4*)(qbase + ks * 32 + 8 * hf + 16);
        }
    }

    const v8f zero = {0.f, 0.f, 0.f, 0.f, 0.f, 0.f, 0.f, 0.f};
    v8f oacc[4] = {zero, zero, zero, zero};
    v8f rsum    = zero;

    // prologue: start DMA of tile 0 into buffer 0
    stage_tile(kn, vT, kT[0], vTile[0], bh, 0, tid);

    for (int it = 0; it < NKT; ++it) {
#if USE_ASYNC_LDS
        __builtin_amdgcn_s_wait_asynccnt(0);   // my copies for buf[it&1] landed
#endif
        __syncthreads();                       // everyone's copies landed

        // kick off DMA of the next tile into the other buffer (its last
        // readers finished in the previous iteration, fenced by the barrier above)
        if (it + 1 < NKT)
            stage_tile(kn, vT, kT[(it + 1) & 1], vTile[(it + 1) & 1],
                       bh, (it + 1) * 64, tid);

        const unsigned short* kTile = kT[it & 1];
        const unsigned short* vTl   = vTile[it & 1];

        // S-tiles: qn @ kn^T, clip/NaN->0, exp, row-sum, stage P to LDS
        for (int ns = 0; ns < 4; ++ns) {
            v8f sacc = zero;
            for (int ks = 0; ks < 2; ++ks) {
                Frag b;
                b.q[0] = *(const uint4*)&kTile[(ns * 16 + ln) * 64 + ks * 32 + 8 * hf];
                b.q[1] = *(const uint4*)&kTile[(ns * 16 + ln) * 64 + ks * 32 + 8 * hf + 16];
                sacc = __builtin_amdgcn_wmma_f32_16x16x32_bf16(
                    false, aq[ks].v, false, b.v, (short)0, sacc, false, false);
            }
            unsigned short* prow = &pT[wave][0];
            for (int r = 0; r < 8; ++r) {
                float x = sacc[r];
                x = (x == x) ? fminf(1.0f, fmaxf(-1.0f, x)) : 0.0f;  // NaN->0, clip
                float p = __expf(x);
                rsum[r] += p;
                prow[(r + 8 * hf) * 64 + ns * 16 + ln] = f2bf(p);
            }
        }

        // O += P @ V  (A = P from LDS bounce, B = v^T tile, both b128 loads)
        for (int ns = 0; ns < 4; ++ns) {
            for (int ks = 0; ks < 2; ++ks) {
                Frag aP, bV;
                aP.q[0] = *(const uint4*)&pT[wave][ln * 64 + ks * 32 + 8 * hf];
                aP.q[1] = *(const uint4*)&pT[wave][ln * 64 + ks * 32 + 8 * hf + 16];
                bV.q[0] = *(const uint4*)&vTl[(ns * 16 + ln) * 64 + ks * 32 + 8 * hf];
                bV.q[1] = *(const uint4*)&vTl[(ns * 16 + ln) * 64 + ks * 32 + 8 * hf + 16];
                oacc[ns] = __builtin_amdgcn_wmma_f32_16x16x32_bf16(
                    false, aP.v, false, bV.v, (short)0, oacc[ns], false, false);
            }
        }
    }

    // softmax denominator: reduce exp-sums over the 16 lanes holding each row
    float rinv[8];
    for (int r = 0; r < 8; ++r) {
        float s = rsum[r];
        s += __shfl_xor(s, 1, 32);
        s += __shfl_xor(s, 2, 32);
        s += __shfl_xor(s, 4, 32);
        s += __shfl_xor(s, 8, 32);
        rinv[r] = 1.0f / s;
    }
    float* obase = out + ((size_t)(bh * S_LEN + q0 + wave * 16)) * HD;
    for (int ns = 0; ns < 4; ++ns)
        for (int r = 0; r < 8; ++r)
            obase[(r + 8 * hf) * HD + ns * 16 + ln] = oacc[ns][r] * rinv[r];
}

// ---------------------------------------------------------------------------
extern "C" void kernel_launch(void* const* d_in, const int* in_sizes, int n_in,
                              void* d_out, int out_size, void* d_ws, size_t ws_size,
                              hipStream_t stream)
{
    (void)in_sizes; (void)n_in; (void)out_size; (void)ws_size;
    const float* query = (const float*)d_in[0];   // [T,B,D] fp32
    const float* W     = (const float*)d_in[1];   // [3D,D]  fp32
    const float* bias  = (const float*)d_in[2];   // [3D]    fp32
    float*       out   = (float*)d_out;           // [B,H,S,hd] fp32

    char* ws = (char*)d_ws;
    const size_t planeBytes = (size_t)BH * S_LEN * HD * sizeof(unsigned short); // 8 MB
    unsigned short* qc = (unsigned short*)(ws);
    unsigned short* kc = (unsigned short*)(ws + planeBytes);
    unsigned short* vT = (unsigned short*)(ws + 2 * planeBytes);

    dim3 g1(ROWS / 64, (3 * DMODEL) / 64);
    qkv_gemm_kernel<<<g1, 128, 0, stream>>>(query, W, bias, qc, kc, vT);
    normalize_kernel<<<(BH * S_LEN) / 256, 256, 0, stream>>>(qc, kc);
    attn_kernel<<<BH * (S_LEN / 64), 128, 0, stream>>>(qc, kc, vT, out);
}
